// MorletWaveletBank_81003083203046
// MI455X (gfx1250) — matrix operand order repaired and checked by reading, not compiled, not run on hardware
//
#include <hip/hip_runtime.h>
#include <hip/hip_bf16.h>

typedef __attribute__((ext_vector_type(16))) _Float16 v16h;
typedef __attribute__((ext_vector_type(8)))  _Float16 v8h;
typedef __attribute__((ext_vector_type(8)))  float    v8f;

#define SIG_LEN   65536
#define NSCALES   32
#define FLEN      256
#define WG_T      256                 // time points per workgroup
#define CHUNKS    (SIG_LEN / WG_T)    // 256
#define NCOPY     8                   // shift-replicated f16 copies
#define CPITCH    528                 // halves per copy row (1056 B, 16B-aligned)

// ---------------------------------------------------------------------------
// Setup: bake the Morlet wavelet bank directly into WMMA A-fragment layout.
// A-fragment k order used here (16-bit A 16x32, wave32, contiguous, matches B):
//   lane<16 : M=lane,    K = 0..15
//   lane>=16: M=lane-16, K = 16..31
//   half j: K = 2*base+j  (base=(lane>=16)*8)
// ws layout: aFrag[ft(4)][kc(8)][lane(32)][j(16)]  (f16)  -> 32 KB
// ---------------------------------------------------------------------------
__global__ void MorletBankPack_kernel(_Float16* __restrict__ aFrag) {
    int idx = blockIdx.x * blockDim.x + threadIdx.x;   // 0..16383
    if (idx >= 4 * 8 * 32 * 16) return;
    int j    =  idx        & 15;
    int lane = (idx >> 4)  & 31;
    int kc   = (idx >> 9)  & 7;
    int ft   = (idx >> 12) & 3;

    int base = (lane >> 4) * 8;
    int k = kc * 32 + base * 2 + j;   // tap index 0..255
    int f = ft * 16 + (lane & 15);                     // 0..31 real, 32..63 imag

    int s_idx   = f & 31;
    float scale = 1.0f + (float)s_idx;                 // linspace(1,32,32)
    float tw    = ((float)k - 128.0f) / scale;
    float env   = __expf(-0.5f * tw * tw);
    float phase = 2.0f * 3.14159265358979323846f * 6.0f * tw / scale;
    float val   = env * ((f < 32) ? __cosf(phase) : __sinf(phase));
    aFrag[idx] = (_Float16)val;
}

// ---------------------------------------------------------------------------
// Main CWT kernel: im2col GEMM via v_wmma_f32_16x16x32_f16.
//   D[f][t] = sum_k W[f][k] * x[t + k - 128]
// A = filters (prepacked fragments in d_ws, aligned coalesced b128, L2-hot)
// B = signal: window converted f32->f16 ONCE, stored as 8 shift-replicated
//     LDS copies so every fragment is two aligned ds_load_b128 (no per-k cvt)
// C/D: time across lanes -> coalesced float2 (complex) stores.
// __launch_bounds__(256,4): keep <=256 VGPRs/wave -> no s_set_vgpr_msb churn,
// 4 waves/SIMD occupancy.
// ---------------------------------------------------------------------------
__global__ __launch_bounds__(256, 4)
void MorletWaveletBank_81003083203046_kernel(const float* __restrict__ sig,
                                             const _Float16* __restrict__ aFrag,
                                             float2* __restrict__ out) {
    // copy s holds window[i+s] at element i; row pitch 528 halves = 1056 B (16B mult)
    __shared__ _Float16 sigh[NCOPY][CPITCH];           // 8448 B

    const int tid   = threadIdx.x;
    const int b     = blockIdx.x >> 8;                 // batch 0..15
    const int chunk = blockIdx.x & (CHUNKS - 1);
    const int wg_t0 = chunk << 8;

    // Stage window [wg_t0-128, wg_t0+383) : 511 samples, cvt once, replicate x8
    const long sigBase = (long)b * SIG_LEN;
    for (int i = tid; i < 511; i += 256) {
        int g = wg_t0 - 128 + i;
        float v = 0.0f;
        if (g >= 0 && g < SIG_LEN) v = sig[sigBase + g];
        _Float16 h = (_Float16)v;
        #pragma unroll
        for (int s = 0; s < NCOPY; ++s) {
            int ii = i - s;
            if (ii >= 0) sigh[s][ii] = h;
        }
    }
    __syncthreads();

    const int wave = tid >> 5;       // 8 waves
    const int lane = tid & 31;
    const int lhi  = (lane >> 4) & 1;
    const int llo  = lane & 15;

    v8f acc[4][2];
    #pragma unroll
    for (int ft = 0; ft < 4; ++ft)
        #pragma unroll
        for (int tt = 0; tt < 2; ++tt)
            acc[ft][tt] = (v8f){0.f, 0.f, 0.f, 0.f, 0.f, 0.f, 0.f, 0.f};

    const v16h* __restrict__ aP = (const v16h*)aFrag;

    #pragma unroll 4
    for (int kc = 0; kc < 8; ++kc) {
        // A fragments: 4 filter tiles, 32 B/lane, aligned + coalesced, L0/L2-hot
        v16h aF[4];
        #pragma unroll
        for (int ft = 0; ft < 4; ++ft)
            aF[ft] = aP[(ft * 8 + kc) * 32 + lane];

        // B fragments: lane = time column; 16 consecutive taps per lane
        //   (lanes 0-15: K 0..15, lanes 16-31: K 16..31 of this chunk)
        // window index q -> copy s=q&7, element a=q>>3: two aligned v8h loads
        v16h bF[2];
        #pragma unroll
        for (int tt = 0; tt < 2; ++tt) {
            const int q = wave * 32 + tt * 16 + llo + kc * 32 + lhi * 16;
            const int s = q & 7;
            const int a = q >> 3;
            const v8h* cp = (const v8h*)(&sigh[s][0]);
            v8h lo = cp[a];
            v8h hi = cp[a + 1];
            bF[tt] = __builtin_shufflevector(lo, hi,
                        0, 1, 2, 3, 4, 5, 6, 7, 8, 9, 10, 11, 12, 13, 14, 15);
        }

        #pragma unroll
        for (int ft = 0; ft < 4; ++ft)
            #pragma unroll
            for (int tt = 0; tt < 2; ++tt)
                acc[ft][tt] = __builtin_amdgcn_wmma_f32_16x16x32_f16(
                    false, aF[ft], false, bF[tt],
                    (short)0, acc[ft][tt], false, false);
    }

    // Store complex64: ft 0,1 = real(scale 0..31), ft 2,3 = imag(scale 0..31).
    // C/D layout: N(=t) = lane&15 across lanes, M(=scale) = r + 8*(lane>=16).
    // Each half-wave writes 16 consecutive t as float2 -> 128 B contiguous.
    #pragma unroll
    for (int tt = 0; tt < 2; ++tt) {
        const int t = wg_t0 + wave * 32 + tt * 16 + llo;
        #pragma unroll
        for (int st = 0; st < 2; ++st) {
            #pragma unroll
            for (int r = 0; r < 8; ++r) {
                const int s = st * 16 + lhi * 8 + r;     // scale 0..31
                const size_t idx = ((size_t)(b * NSCALES + s)) * SIG_LEN + t;
                float2 v;
                v.x = acc[st][tt][r];       // real
                v.y = acc[st + 2][tt][r];   // imag
                out[idx] = v;
            }
        }
    }
}

extern "C" void kernel_launch(void* const* d_in, const int* in_sizes, int n_in,
                              void* d_out, int out_size, void* d_ws, size_t ws_size,
                              hipStream_t stream) {
    (void)in_sizes; (void)n_in; (void)out_size; (void)ws_size;

    const float* sig   = (const float*)d_in[0];
    _Float16*    aFrag = (_Float16*)d_ws;          // 32 KB of fragments
    float2*      out   = (float2*)d_out;           // complex64 interleaved

    // Bake wavelet bank into WMMA A-fragment layout (4*8*32*16 = 16384 f16)
    MorletBankPack_kernel<<<64, 256, 0, stream>>>(aFrag);

    // 16 batches x 256 time-chunks, 8 waves/block
    MorletWaveletBank_81003083203046_kernel<<<16 * CHUNKS, 256, 0, stream>>>(sig, aFrag, out);
}
